// EquivariantActivationModule_65309272703460
// MI455X (gfx1250) — matrix operand order repaired and checked by reading, not compiled, not run on hardware
//
#include <hip/hip_runtime.h>

typedef __bf16 bf16_t;
typedef __attribute__((ext_vector_type(8)))  __bf16 bf16x8;
typedef __attribute__((ext_vector_type(16))) __bf16 bf16x16;
typedef __attribute__((ext_vector_type(8)))  float  f32x8;

#define M_    8192          // B*N points
#define F_    8192          // 2*C*C features (K of the GEMM)
#define U_IN  512
#define NCOL  2560          // 512 + 1024 + 1024 output columns
#define WLC   2048          // equivariant activation columns kept in ws

#define BM 128
#define BN 128
#define BK 64
#define NSTEP (F_ / BK)     // 128
#define LDT 72              // padded LDS stride in bf16 (144 B, keeps 16B-aligned chunks)

#define CAT16(lo,hi) __builtin_shufflevector(lo, hi, 0,1,2,3,4,5,6,7,8,9,10,11,12,13,14,15)

// ---------------------------------------------------------------------------
// Kernel 1: weight transpose + fp32 -> bf16 convert via 32x32 LDS tile.
// Source: W_inv (8192,512), W_eq (2,8192,1024).  Dest: Wt[n][k] N-major.
// ---------------------------------------------------------------------------
__global__ __launch_bounds__(256) void wconv_kernel(const float* __restrict__ W_inv,
                                                    const float* __restrict__ W_eq,
                                                    bf16_t* __restrict__ Wt) {
    __shared__ float tile[32][33];
    const int k0 = (int)blockIdx.x * 32;
    const int n0 = (int)blockIdx.y * 32;
    const int tx = threadIdx.x & 31;        // fast index
    const int ty = threadIdx.x >> 5;        // 0..7
    #pragma unroll
    for (int i = 0; i < 4; ++i) {           // coalesced read along n
        int k = k0 + ty + 8 * i;
        int n = n0 + tx;
        float v;
        if (n < U_IN) {
            v = W_inv[(size_t)k * U_IN + n];
        } else {
            int j = n - U_IN;
            int l = j >> 10, jj = j & 1023;
            v = W_eq[((size_t)l * F_ + k) * 1024 + jj];
        }
        tile[ty + 8 * i][tx] = v;
    }
    __syncthreads();
    #pragma unroll
    for (int i = 0; i < 4; ++i) {           // coalesced bf16 write along k
        int n = n0 + ty + 8 * i;
        int k = k0 + tx;
        Wt[(size_t)n * F_ + k] = (bf16_t)tile[tx][ty + 8 * i];
    }
}

// ---------------------------------------------------------------------------
// Kernel 2: Gram features per point, l2-norm over D folded as column scale.
// ---------------------------------------------------------------------------
__global__ __launch_bounds__(256) void feats_kernel(const float* __restrict__ t1,
                                                    const float* __restrict__ t2,
                                                    const float* __restrict__ u1,
                                                    const float* __restrict__ u2,
                                                    bf16_t* __restrict__ feats) {
    const int row = blockIdx.x;
    __shared__ float sT[2 * 192];
    __shared__ float sU[2 * 192];
    __shared__ float sRN[128];
    const float* tsrc[2] = { t1, t2 };
    const float* usrc[2] = { u1, u2 };
    for (int i = threadIdx.x; i < 384; i += 256) {
        int hh = i / 192, r = i % 192;
        sT[i] = tsrc[hh][(size_t)row * 192 + r];
        sU[i] = usrc[hh][(size_t)row * 192 + r];
    }
    __syncthreads();
    for (int i = threadIdx.x; i < 128; i += 256) {
        int hh = i >> 6, c = i & 63;
        const float* ub = &sU[hh * 192 + c];
        float s = ub[0] * ub[0] + ub[64] * ub[64] + ub[128] * ub[128];
        sRN[i] = rsqrtf(fmaxf(s, 0.01f));
    }
    __syncthreads();
    bf16_t* dst = feats + (size_t)row * F_;
    #pragma unroll 4
    for (int k = 0; k < 32; ++k) {
        int f  = k * 256 + threadIdx.x;          // coalesced along f
        int hh = f >> 12;
        int p  = f & 4095;
        int i  = p >> 6, j = p & 63;
        const float* tb = &sT[hh * 192];
        const float* ub = &sU[hh * 192];
        float v = (tb[i] * ub[j] + tb[64 + i] * ub[64 + j] + tb[128 + i] * ub[128 + j])
                  * sRN[hh * 64 + j];
        dst[f] = (bf16_t)v;
    }
}

// ---------------------------------------------------------------------------
// Kernel 3: bf16 WMMA GEMM (M=8192, K=8192, N=2560) + BN + ReLU epilogue.
// 128x128 block tile, 8 waves, wave tile 64x32 (4x2 WMMA tiles), K-step 64.
// Double-buffered LDS, register-staged global loads for overlap.
// ---------------------------------------------------------------------------
__global__ __launch_bounds__(256) void gemm_bn_relu(
        const bf16_t* __restrict__ A,      // feats, row-major MxK
        const bf16_t* __restrict__ Bw,     // Wt, N-major NCOLxK
        const float* __restrict__ b_inv, const float* __restrict__ g_inv,
        const float* __restrict__ be_inv,
        const float* __restrict__ b_eq,  const float* __restrict__ g_eq,
        const float* __restrict__ be_eq,
        float* __restrict__ invar_out,     // (M, 512) part of d_out
        float* __restrict__ wl_out) {      // (M, 2048) ws
    __shared__ bf16_t sA[2][BM * LDT];
    __shared__ bf16_t sB[2][BN * LDT];

    const int tid    = threadIdx.x;
    const int lane   = tid & 31;
    const int wave   = tid >> 5;
    const int waveM  = wave >> 2;                 // 0..1
    const int waveN  = wave & 3;                  // 0..3
    const int blockM = (int)blockIdx.y * BM;
    const int blockN = (int)blockIdx.x * BN;
    const int h  = lane >> 4;                     // half-wave select
    const int lr = lane & 15;

    // Staging map: 128 rows x 8 chunks(8 bf16) per matrix; thread s-loop covers 4.
    const int srow = tid >> 3;                    // 0..31
    const int skc  = (tid & 7) * 8;               // 0,8,..,56
    const bf16_t* gA = A  + (size_t)(blockM + srow) * F_ + skc;
    const bf16_t* gB = Bw + (size_t)(blockN + srow) * F_ + skc;
    const int sbase = srow * LDT + skc;

    bf16x8 stA[4], stB[4];
    #pragma unroll
    for (int s = 0; s < 4; ++s) {                 // preload tile 0
        stA[s] = *(const bf16x8*)(gA + (size_t)s * 32 * F_);
        stB[s] = *(const bf16x8*)(gB + (size_t)s * 32 * F_);
    }

    f32x8 acc[4][2] = {};
    int p = 0;
    for (int kb = 0; kb < NSTEP; ++kb) {
        #pragma unroll
        for (int s = 0; s < 4; ++s) {             // commit staged tile to LDS
            *(bf16x8*)&sA[p][sbase + s * 32 * LDT] = stA[s];
            *(bf16x8*)&sB[p][sbase + s * 32 * LDT] = stB[s];
        }
        __syncthreads();

        // Branch-free next-tile fetch (last iteration wraps to 0, never stored).
        const int knext = ((kb + 1) & (NSTEP - 1)) * BK;
        #pragma unroll
        for (int s = 0; s < 4; ++s) {
            stA[s] = *(const bf16x8*)(gA + (size_t)s * 32 * F_ + knext);
            stB[s] = *(const bf16x8*)(gB + (size_t)s * 32 * F_ + knext);
        }

        // Compute on buffer p while next-tile global loads are in flight.
        #pragma unroll
        for (int kk = 0; kk < BK; kk += 32) {
            bf16x16 af[4];
            #pragma unroll
            for (int mi = 0; mi < 4; ++mi) {
                int r = waveM * 64 + mi * 16 + lr;
                bf16x8 lo = *(const bf16x8*)&sA[p][r * LDT + kk + 8 * h];
                bf16x8 hi = *(const bf16x8*)&sA[p][r * LDT + kk + 16 + 8 * h];
                af[mi] = CAT16(lo, hi);
            }
            bf16x16 bfr[2];
            #pragma unroll
            for (int nj = 0; nj < 2; ++nj) {
                int n = waveN * 32 + nj * 16 + lr;
                bf16x8 lo = *(const bf16x8*)&sB[p][n * LDT + kk + 8 * h];
                bf16x8 hi = *(const bf16x8*)&sB[p][n * LDT + kk + 16 + 8 * h];
                bfr[nj] = CAT16(lo, hi);
            }
            #pragma unroll
            for (int mi = 0; mi < 4; ++mi)
                #pragma unroll
                for (int nj = 0; nj < 2; ++nj)
                    acc[mi][nj] = __builtin_amdgcn_wmma_f32_16x16x32_bf16(
                        false, af[mi], false, bfr[nj], (short)0, acc[mi][nj], false, false);
        }
        p ^= 1;
    }

    // Epilogue: relu(gamma*rsqrt(1.001)*(x + b) + beta)
    const float RSQ = rsqrtf(1.001f);
    #pragma unroll
    for (int nj = 0; nj < 2; ++nj) {
        int c = blockN + waveN * 32 + nj * 16 + lr;
        float sc, bia, bet;
        if (c < U_IN) { sc = g_inv[c] * RSQ; bia = b_inv[c]; bet = be_inv[c]; }
        else { int j = c - U_IN; sc = g_eq[j] * RSQ; bia = b_eq[j]; bet = be_eq[j]; }
        #pragma unroll
        for (int mi = 0; mi < 4; ++mi) {
            int rbase = blockM + waveM * 64 + mi * 16 + h * 8;
            f32x8 a = acc[mi][nj];
            #pragma unroll
            for (int v = 0; v < 8; ++v) {
                float val = sc * (a[v] + bia) + bet;
                val = val > 0.0f ? val : 0.0f;
                int r = rbase + v;
                if (c < U_IN) invar_out[(size_t)r * U_IN + c] = val;
                else          wl_out[(size_t)r * WLC + (c - U_IN)] = val;
            }
        }
    }
}

// ---------------------------------------------------------------------------
// Kernel 4: basis recombine  eq[b,n,m, l*64+i] = sum_j wl[row, l*1024+i*16+j]*V_l[row,m,j]
// ---------------------------------------------------------------------------
__global__ void recomb_kernel(const float* __restrict__ wl,
                              const float* __restrict__ V1,
                              const float* __restrict__ V2,
                              float* __restrict__ out_eq) {
    long e = (long)blockIdx.x * 256 + threadIdx.x;
    if (e >= (long)M_ * 3 * 128) return;
    int row = (int)(e / 384);
    int rem = (int)(e % 384);
    int m  = rem >> 7;          // 0..2
    int cc = rem & 127;         // 0..127
    int l  = cc >> 6;           // head
    int i  = cc & 63;           // unit
    const float* V = l ? V2 : V1;
    const float* w = wl + (size_t)row * WLC + l * 1024 + i * 16;
    const float* v = V + (size_t)row * 48 + m * 16;
    float s = 0.0f;
    #pragma unroll
    for (int j = 0; j < 16; ++j) s += w[j] * v[j];
    out_eq[e] = s;
}

// ---------------------------------------------------------------------------
extern "C" void kernel_launch(void* const* d_in, const int* in_sizes, int n_in,
                              void* d_out, int out_size, void* d_ws, size_t ws_size,
                              hipStream_t stream) {
    const float* t1    = (const float*)d_in[0];
    const float* t2    = (const float*)d_in[1];
    const float* u1    = (const float*)d_in[2];
    const float* u2    = (const float*)d_in[3];
    const float* V1    = (const float*)d_in[4];
    const float* V2    = (const float*)d_in[5];
    const float* W_inv = (const float*)d_in[6];
    const float* b_inv = (const float*)d_in[7];
    const float* g_inv = (const float*)d_in[8];
    const float* be_inv= (const float*)d_in[9];
    const float* W_eq  = (const float*)d_in[10];
    const float* b_eq  = (const float*)d_in[11];
    const float* g_eq  = (const float*)d_in[12];
    const float* be_eq = (const float*)d_in[13];

    char* ws = (char*)d_ws;
    const size_t WT_BYTES    = (size_t)NCOL * F_ * sizeof(bf16_t);   // 41,943,040
    const size_t FEATS_BYTES = (size_t)M_ * F_ * sizeof(bf16_t);     // 134,217,728
    bf16_t* Wt    = (bf16_t*)ws;
    bf16_t* feats = (bf16_t*)(ws + WT_BYTES);
    float*  wl    = (float*)(ws + WT_BYTES + FEATS_BYTES);

    float* out    = (float*)d_out;
    float* out_eq = out + (size_t)M_ * U_IN;

    dim3 gridW(F_ / 32, NCOL / 32);
    wconv_kernel<<<gridW, 256, 0, stream>>>(W_inv, W_eq, Wt);
    feats_kernel<<<M_, 256, 0, stream>>>(t1, t2, u1, u2, feats);
    dim3 gridG(NCOL / BN, M_ / BM);
    gemm_bn_relu<<<gridG, 256, 0, stream>>>(feats, Wt,
                                            b_inv, g_inv, be_inv,
                                            b_eq, g_eq, be_eq,
                                            out, wl);
    recomb_kernel<<<((long)M_ * 384) / 256, 256, 0, stream>>>(wl, V1, V2, out_eq);
}
